// DRMM_87479893885328
// MI455X (gfx1250) — matrix operand order repaired and checked by reading, not compiled, not run on hardware
//
#include <hip/hip_runtime.h>
#include <hip/hip_bf16.h>
#include <math.h>

// ---------------------------------------------------------------------------
// DRMM forward for MI455X (gfx1250, wave32, WMMA).
//   B=32, Lq=16, Ld=2048, V=50000, E=300 (padded to 320), BINS=30
// Bandwidth-bound: ~79MB embedding gather dominates; cos GEMM done with
// v_wmma_f32_16x16x32_bf16 on bf16-staged normalized embeddings.
// ---------------------------------------------------------------------------

typedef __attribute__((ext_vector_type(16))) __bf16 v16bf;
typedef __attribute__((ext_vector_type(8)))  __bf16 v8bf;
typedef __attribute__((ext_vector_type(8)))  float  v8f;

#define NB    32      // batch
#define LQ    16      // query len  (== WMMA M)
#define LD    2048    // doc len
#define EDIM  300
#define KPAD  320     // 10 * 32 (WMMA K steps)
#define BINS  30
#define TINY  1e-13f

// ---------------------------------------------------------------------------
// Kernel 0: zero the global histogram accumulator.
// ---------------------------------------------------------------------------
__global__ void drmm_zero_hist(float* __restrict__ hist, int n) {
    int i = blockIdx.x * blockDim.x + threadIdx.x;
    if (i < n) hist[i] = 0.0f;
}

// ---------------------------------------------------------------------------
// Kernel 1/2: gather embedding row, apply mask, L2-normalize, emit bf16 row
// padded to KPAD. One wave32 per row; 8 waves per 256-thread block.
// ---------------------------------------------------------------------------
__global__ void drmm_normalize_rows(const int*   __restrict__ ids,
                                    const float* __restrict__ mask,
                                    const float* __restrict__ emb,
                                    __bf16*      __restrict__ out,
                                    int nrows) {
    const int wave = threadIdx.x >> 5;
    const int lane = threadIdx.x & 31;
    const int row  = blockIdx.x * (blockDim.x >> 5) + wave;
    if (row >= nrows) return;

    const int   id = ids[row];
    const float m  = mask[row];
    const float* __restrict__ src = emb + (long long)id * EDIM;

    float v[10];
    float ssq = 0.0f;
#pragma unroll
    for (int i = 0; i < 10; ++i) {
        const int k = lane + 32 * i;                 // covers [0, 320)
        const float x = (k < EDIM) ? src[k] * m : 0.0f;
        v[i] = x;
        ssq += x * x;
    }
    // wave32 reduction
#pragma unroll
    for (int off = 16; off > 0; off >>= 1)
        ssq += __shfl_xor(ssq, off, 32);

    const float scale = 1.0f / (sqrtf(ssq) + TINY);
    __bf16* __restrict__ dst = out + (long long)row * KPAD;
#pragma unroll
    for (int i = 0; i < 10; ++i) {
        const int k = lane + 32 * i;
        dst[k] = (__bf16)(v[i] * scale);
    }
}

// ---------------------------------------------------------------------------
// Kernel 3: cos = Qn · Dnᵀ per 16x16 tile via v_wmma_f32_16x16x32_bf16,
// fused histogram into LDS (ds_add_f32), one global-atomic flush per block.
// grid = (LD/16/4, NB), block = 128 (4 waves; wave w owns n-tile 4*bx+w).
// ---------------------------------------------------------------------------
__global__ void drmm_cos_hist(const __bf16* __restrict__ Qn,   // [NB][16][KPAD]
                              const __bf16* __restrict__ Dn,   // [NB][LD][KPAD]
                              float*        __restrict__ hist) // [NB][16][BINS]
{
    __shared__ float hist_s[LQ * 32];          // 16 rows x 32 (bins padded)

    const int tid  = threadIdx.x;
    const int b    = blockIdx.y;
    const int wave = tid >> 5;
    const int lane = tid & 31;
    const int l16  = lane & 15;
    const bool hi  = lane >= 16;

    for (int i = tid; i < LQ * 32; i += 128) hist_s[i] = 0.0f;
    __syncthreads();

    const int ntile = blockIdx.x * 4 + wave;   // 0..127

    // A fragment base: row M = l16; K-halves offset by 8 for upper lanes
    const __bf16* __restrict__ abase = Qn + ((long long)b * LQ + l16) * KPAD + (hi ? 8 : 0);
    // B fragment base: doc row N = ntile*16 + l16; K-halves offset by 8 for upper lanes
    const __bf16* __restrict__ bbase = Dn + ((long long)b * LD + ntile * 16 + l16) * KPAD + (hi ? 8 : 0);

    v8f acc = {};
#pragma unroll
    for (int kk = 0; kk < KPAD / 32; ++kk) {
        const int k0 = kk * 32;
        const v8bf a_lo = *(const v8bf*)(abase + k0);        // K = k0+{0..7}(+8)
        const v8bf a_hi = *(const v8bf*)(abase + k0 + 16);   // K = k0+{16..23}(+8)
        v16bf a;
#pragma unroll
        for (int i = 0; i < 8; ++i) { a[i] = a_lo[i]; a[i + 8] = a_hi[i]; }
        const v8bf b_lo = *(const v8bf*)(bbase + k0);        // K = k0+{0..7}(+8)
        const v8bf b_hi = *(const v8bf*)(bbase + k0 + 16);   // K = k0+{16..23}(+8)
        v16bf bb;
#pragma unroll
        for (int i = 0; i < 8; ++i) { bb[i] = b_lo[i]; bb[i + 8] = b_hi[i]; }
        acc = __builtin_amdgcn_wmma_f32_16x16x32_bf16(
            /*neg_a=*/false, a, /*neg_b=*/false, bb,
            /*c_mod=*/(short)0, acc, /*reuse_a=*/false, /*reuse_b=*/false);
    }

    // D layout: VGPR i, lanes 0-15 -> M=i, lanes 16-31 -> M=i+8.
#pragma unroll
    for (int i = 0; i < 8; ++i) {
        const int M = hi ? (i + 8) : i;
        const float c = acc[i];
        int bin = (int)floorf((c + 1.0f) * (BINS * 0.5f));
        bin = bin < 0 ? 0 : (bin > BINS - 1 ? BINS - 1 : bin);
        atomicAdd(&hist_s[M * 32 + bin], 1.0f);             // ds_add_f32
    }
    __syncthreads();

    float* __restrict__ hg = hist + (long long)b * LQ * BINS;
    for (int i = tid; i < LQ * BINS; i += 128) {
        const int q = i / BINS, bin = i % BINS;
        const float v = hist_s[q * 32 + bin];
        if (v != 0.0f) unsafeAtomicAdd(&hg[q * BINS + bin], v);
    }
}

// ---------------------------------------------------------------------------
// Kernel 4: per-batch head. Match MLP on log1p(hist), gate MLP on raw masked
// q_emb, masked softmax over q, sigmoid. One 128-thread block per batch.
// ---------------------------------------------------------------------------
__global__ void drmm_head(const int*   __restrict__ q_ids,
                          const float* __restrict__ q_mask,
                          const float* __restrict__ emb,
                          const float* __restrict__ Wm1, const float* __restrict__ bm1,
                          const float* __restrict__ Wm2, const float* __restrict__ bm2,
                          const float* __restrict__ Wg1, const float* __restrict__ bg1,
                          const float* __restrict__ Wg2, const float* __restrict__ bg2,
                          const float* __restrict__ hist,
                          float*       __restrict__ out)
{
    __shared__ float qe[EDIM];
    __shared__ float red[128];
    __shared__ float hmlp[BINS];
    __shared__ float match_s[LQ];
    __shared__ float g_s[LQ];

    const int b = blockIdx.x;
    const int t = threadIdx.x;

    for (int q = 0; q < LQ; ++q) {
        const int   id = q_ids[b * LQ + q];
        const float m  = q_mask[b * LQ + q];
        for (int k = t; k < EDIM; k += 128) qe[k] = emb[(long long)id * EDIM + k] * m;
        __syncthreads();

        // gate net: sum_e' Wg2[e'] * tanh(Wg1[e',:]·qe + bg1[e'])
        float part = 0.0f;
        for (int ep = t; ep < EDIM; ep += 128) {
            const float* __restrict__ wrow = Wg1 + (long long)ep * EDIM;
            float s = bg1[ep];
            for (int k = 0; k < EDIM; ++k) s = fmaf(wrow[k], qe[k], s);
            part += Wg2[ep] * tanhf(s);
        }
        red[t] = part;
        __syncthreads();
        for (int s = 64; s > 0; s >>= 1) {
            if (t < s) red[t] += red[t + s];
            __syncthreads();
        }

        // match net: tanh(Wm2 · tanh(Wm1 · log1p(hist[b,q]) + bm1) + bm2)
        if (t < BINS) {
            const float* __restrict__ hrow = hist + ((long long)b * LQ + q) * BINS;
            const float* __restrict__ wrow = Wm1 + t * BINS;
            float s = bm1[t];
            for (int j = 0; j < BINS; ++j) s = fmaf(wrow[j], log1pf(hrow[j]), s);
            hmlp[t] = tanhf(s);
        }
        __syncthreads();
        if (t == 0) {
            float s = bm2[0];
            for (int j = 0; j < BINS; ++j) s = fmaf(Wm2[j], hmlp[j], s);
            match_s[q] = tanhf(s);
            g_s[q]     = tanhf(red[0] + bg2[0]);
        }
        __syncthreads();
    }

    if (t == 0) {
        // masked softmax over q (mask=0 -> -inf before max), then score
        float mx = -INFINITY;
        for (int q = 0; q < LQ; ++q) {
            const float mk = q_mask[b * LQ + q];
            const float xm = (mk > 0.0f) ? g_s[q] : -INFINITY;
            mx = fmaxf(mx, xm);
        }
        float e[LQ], sum = 0.0f;
        for (int q = 0; q < LQ; ++q) {
            const float mk = q_mask[b * LQ + q];
            e[q] = expf(g_s[q] - mx) * mk;
            sum += e[q];
        }
        float sc = 0.0f;
        for (int q = 0; q < LQ; ++q) sc += match_s[q] * (e[q] / sum);
        out[b] = 1.0f / (1.0f + expf(-sc));
    }
}

// ---------------------------------------------------------------------------
// Host-side launch
// ---------------------------------------------------------------------------
extern "C" void kernel_launch(void* const* d_in, const int* in_sizes, int n_in,
                              void* d_out, int out_size, void* d_ws, size_t ws_size,
                              hipStream_t stream) {
    const int*   q_ids  = (const int*)  d_in[0];
    const float* q_mask = (const float*)d_in[1];
    const int*   d_ids  = (const int*)  d_in[2];
    const float* d_mask = (const float*)d_in[3];
    const float* emb    = (const float*)d_in[4];
    const float* Wm1    = (const float*)d_in[5];
    const float* bm1    = (const float*)d_in[6];
    const float* Wm2    = (const float*)d_in[7];
    const float* bm2    = (const float*)d_in[8];
    const float* Wg1    = (const float*)d_in[9];
    const float* bg1    = (const float*)d_in[10];
    const float* Wg2    = (const float*)d_in[11];
    const float* bg2    = (const float*)d_in[12];
    float* out = (float*)d_out;

    // workspace layout (256B-aligned segments)
    char* ws = (char*)d_ws;
    __bf16* Dn   = (__bf16*)(ws);                                      // 32*2048*320 bf16
    __bf16* Qn   = (__bf16*)(ws + (size_t)NB * LD * KPAD * 2);         // 32*16*320 bf16
    float*  hist = (float*) (ws + (size_t)NB * LD * KPAD * 2
                                 + (size_t)NB * LQ * KPAD * 2);        // 32*16*30 f32

    // 0) zero histogram (poisoned workspace)
    const int hist_n = NB * LQ * BINS;
    drmm_zero_hist<<<(hist_n + 255) / 256, 256, 0, stream>>>(hist, hist_n);

    // 1) docs: 65536 rows, 1 wave/row, 8 waves/block
    drmm_normalize_rows<<<(NB * LD) / 8, 256, 0, stream>>>(d_ids, d_mask, emb, Dn, NB * LD);

    // 2) queries: 512 rows
    drmm_normalize_rows<<<(NB * LQ) / 8, 256, 0, stream>>>(q_ids, q_mask, emb, Qn, NB * LQ);

    // 3) WMMA cosine + fused histogram: grid (LD/16/4, NB), 4 waves/block
    drmm_cos_hist<<<dim3(LD / 16 / 4, NB), 128, 0, stream>>>(Qn, Dn, hist);

    // 4) head: one block per batch
    drmm_head<<<NB, 128, 0, stream>>>(q_ids, q_mask, emb,
                                      Wm1, bm1, Wm2, bm2,
                                      Wg1, bg1, Wg2, bg2,
                                      hist, out);
}
